// GreedyRNNTV2_84344567759684
// MI455X (gfx1250) — matrix operand order repaired and checked by reading, not compiled, not run on hardware
//
#include <hip/hip_runtime.h>
#include <stdint.h>

// Problem constants (from reference)
#define Bsz    64
#define Tt     512
#define DENCc  1024
#define Ee     640
#define Hh     640
#define Jj     640
#define Vv     4096
#define MAXSYM 3
#define OUTW   (Tt*MAXSYM + 1)   // 1537

// Persistent-kernel shape
#define NBLK   64
#define NTHR   256
#define NWAVE  ((NBLK*NTHR)/32)  // 512 waves

typedef __attribute__((ext_vector_type(16))) __bf16 v16bf;
typedef __attribute__((ext_vector_type(8)))  __bf16 v8bf;
typedef __attribute__((ext_vector_type(8)))  float  v8f;

#define WMMA_BF16(acc, a, b) \
  __builtin_amdgcn_wmma_f32_16x16x32_bf16(false, (a), false, (b), (short)0, (acc), false, false)

// ---------------- WMMA fragment loaders (wave32 layouts per CDNA5 ISA 7.12.2) --
// A 16x32 (MxK): lanes 0-15: M=lane, K={0..7,16..23}; lanes 16-31: M=lane-16, K={8..15,24..31}
// B 32x16 (KxN): lanes 0-15: N=lane, K={0..7,16..23}; lanes 16-31: N=lane-16, K={8..15,24..31}
// C/D f32 16x16: lanes 0-15: N=lane rows 0..7; lanes 16-31: N=lane-16 rows 8..15

// A fragment from a per-lane row base pointer (bf16 source), 2x16B vector loads
__device__ inline v16bf load_A_row(const __bf16* rowbase, int kcol, int lane) {
  int k0 = ((lane & 16) ? 8 : 0) + kcol;
  v8bf lo = *(const v8bf*)(rowbase + k0);
  v8bf hi = *(const v8bf*)(rowbase + k0 + 16);
  return __builtin_shufflevector(lo, hi, 0,1,2,3,4,5,6,7,8,9,10,11,12,13,14,15);
}

// A fragment from f32 source rows (encoder activations), 4x16B loads + cvt
__device__ inline v16bf load_A_row_f32(const float* rowbase, int kcol, int lane) {
  int k0 = ((lane & 16) ? 8 : 0) + kcol;
  float4 x0 = *(const float4*)(rowbase + k0);
  float4 x1 = *(const float4*)(rowbase + k0 + 4);
  float4 x2 = *(const float4*)(rowbase + k0 + 16);
  float4 x3 = *(const float4*)(rowbase + k0 + 20);
  v16bf a;
  a[0]=(__bf16)x0.x;  a[1]=(__bf16)x0.y;  a[2]=(__bf16)x0.z;  a[3]=(__bf16)x0.w;
  a[4]=(__bf16)x1.x;  a[5]=(__bf16)x1.y;  a[6]=(__bf16)x1.z;  a[7]=(__bf16)x1.w;
  a[8]=(__bf16)x2.x;  a[9]=(__bf16)x2.y;  a[10]=(__bf16)x2.z; a[11]=(__bf16)x2.w;
  a[12]=(__bf16)x3.x; a[13]=(__bf16)x3.y; a[14]=(__bf16)x3.z; a[15]=(__bf16)x3.w;
  return a;
}

// B fragment from pre-swizzled weights: lane's 16 values are contiguous (32B)
__device__ inline v16bf load_B_swz(const __bf16* base, int tile, int lane) {
  return *(const v16bf*)(base + (((size_t)tile << 5) + (size_t)lane) * 16);
}

__device__ inline void store_C(float* dst, int ld, int lane, const v8f& c) {
  int n  = lane & 15;
  int mo = (lane & 16) ? 8 : 0;
#pragma unroll
  for (int r = 0; r < 8; ++r) dst[(size_t)(mo + r) * ld + n] = c[r];
}

__device__ inline float sigf(float x) { return 1.0f / (1.0f + expf(-x)); }

__device__ inline unsigned long long shfl_xor_u64(unsigned long long v, int m) {
  unsigned lo = (unsigned)(v & 0xFFFFFFFFull);
  unsigned hi = (unsigned)(v >> 32);
  lo = __shfl_xor(lo, m, 32);
  hi = __shfl_xor(hi, m, 32);
  return ((unsigned long long)hi << 32) | (unsigned long long)lo;
}

// ---------------- parameters ------------------------------------------------

struct Params {
  const float* enc;        // [B,T,DENC] f32
  const int*   enc_len;    // [B]
  const float* b_lstm;     // [4H]
  const float* b_joint;    // [J]
  const float* b_out;      // [V]
  const __bf16* embed;     // [V,E] bf16 row-major (A gather source)
  const __bf16* wx;        // swizzled B-fragments of [E,4H]
  const __bf16* wh;        // swizzled [H,4H]
  const __bf16* wenc;      // swizzled [DENC,J]
  const __bf16* wdec;      // swizzled [H,J]
  const __bf16* wout;      // swizzled [J,V]
  float* h; float* c; float* nh; float* nc; float* encp; float* gates;
  __bf16* hb; __bf16* nhb; __bf16* preb;
  int* cur; int* endf;
  unsigned long long* amax;
  int* bar;                // [2]: arrive count, generation
  int* out;                // [B, OUTW] int32
};

// Device-wide software barrier (persistent grid, generation counter)
__device__ inline void gbar(int* bar) {
  __syncthreads();
  if (threadIdx.x == 0) {
    __threadfence();
    int gen = atomicAdd(&bar[1], 0);
    int a   = atomicAdd(&bar[0], 1);
    if (a == NBLK - 1) {
      bar[0] = 0;
      __threadfence();
      atomicAdd(&bar[1], 1);
    } else {
      while (atomicAdd(&bar[1], 0) == gen) __builtin_amdgcn_s_sleep(8);
    }
    __threadfence();
  }
  __syncthreads();
}

__device__ inline void argmax_epilogue(const v8f& acc, int rb, int col, int lane,
                                       const float* b_out, unsigned long long* amax) {
  int mo = (lane & 16) ? 8 : 0;
  float bo = b_out[col];
#pragma unroll
  for (int r = 0; r < 8; ++r) {
    int row = rb * 16 + mo + r;
    float logit = acc[r] + bo;
    unsigned u = __float_as_uint(logit);
    unsigned k32 = (u & 0x80000000u) ? ~u : (u | 0x80000000u);
    unsigned long long key =
        ((unsigned long long)k32 << 32) |
        (unsigned long long)(0xFFFFFFFFu - (unsigned)col);  // ties -> lowest index
#pragma unroll
    for (int m = 1; m < 16; m <<= 1) {
      unsigned long long o = shfl_xor_u64(key, m);
      if (o > key) key = o;
    }
    if ((lane & 15) == 0) atomicMax(&amax[row], key);
  }
}

// ---------------- persistent decode kernel ----------------------------------

__global__ __launch_bounds__(NTHR) void rnnt_greedy_persistent(Params p) {
  const int tid  = threadIdx.x;
  const int gid  = blockIdx.x * NTHR + tid;
  const int lane = tid & 31;
  const int wave = gid >> 5;
  const int nthr = NBLK * NTHR;
  const int m    = lane & 15;

  for (int t = 0; t < Tt; ++t) {
    // ---- Stage 0: enc_proj = enc[:,t,:] @ W_enc  ([64,1024] x [1024,640])
    {
      const int NT = Jj / 16;  // 40 col tiles
      for (int pair = wave; pair < 4 * (NT / 2); pair += NWAVE) {
        int rb = pair & 3, nb0 = (pair >> 2) * 2;
        const float* arow = p.enc + ((size_t)(rb * 16 + m) * Tt + t) * DENCc;
        v8f acc0 = {}, acc1 = {};
        for (int kt = 0; kt < DENCc / 32; ++kt) {
          v16bf a  = load_A_row_f32(arow, kt * 32, lane);
          v16bf b0 = load_B_swz(p.wenc, kt * NT + nb0, lane);
          v16bf b1 = load_B_swz(p.wenc, kt * NT + nb0 + 1, lane);
          acc0 = WMMA_BF16(acc0, a, b0);
          acc1 = WMMA_BF16(acc1, a, b1);
        }
        store_C(p.encp + (rb * 16) * Jj + nb0 * 16, Jj, lane, acc0);
        store_C(p.encp + (rb * 16) * Jj + (nb0 + 1) * 16, Jj, lane, acc1);
        if (t + 1 < Tt) __builtin_prefetch(arow + DENCc, 0, 0);  // next frame slice
      }
    }
    gbar(p.bar);

    for (int s = 0; s < MAXSYM; ++s) {
      // ---- Stage 1: gates = embed[cur] @ Wx + h @ Wh + b_lstm  ([64,640]->[64,2560])
      {
        const int NT = 4 * Hh / 16;  // 160
        for (int pair = wave; pair < 4 * (NT / 2); pair += NWAVE) {
          int rb = pair & 3, nb0 = (pair >> 2) * 2;
          const __bf16* arow_e = p.embed + (size_t)p.cur[rb * 16 + m] * Ee;
          const __bf16* arow_h = p.hb + (size_t)(rb * 16 + m) * Hh;
          v8f acc0 = {}, acc1 = {};
          for (int kt = 0; kt < Ee / 32; ++kt) {
            v16bf a  = load_A_row(arow_e, kt * 32, lane);
            v16bf b0 = load_B_swz(p.wx, kt * NT + nb0, lane);
            v16bf b1 = load_B_swz(p.wx, kt * NT + nb0 + 1, lane);
            acc0 = WMMA_BF16(acc0, a, b0);
            acc1 = WMMA_BF16(acc1, a, b1);
          }
          for (int kt = 0; kt < Hh / 32; ++kt) {
            v16bf a  = load_A_row(arow_h, kt * 32, lane);
            v16bf b0 = load_B_swz(p.wh, kt * NT + nb0, lane);
            v16bf b1 = load_B_swz(p.wh, kt * NT + nb0 + 1, lane);
            acc0 = WMMA_BF16(acc0, a, b0);
            acc1 = WMMA_BF16(acc1, a, b1);
          }
          float bias0 = p.b_lstm[nb0 * 16 + m];
          float bias1 = p.b_lstm[(nb0 + 1) * 16 + m];
#pragma unroll
          for (int r = 0; r < 8; ++r) { acc0[r] += bias0; acc1[r] += bias1; }
          store_C(p.gates + (size_t)(rb * 16) * (4 * Hh) + nb0 * 16, 4 * Hh, lane, acc0);
          store_C(p.gates + (size_t)(rb * 16) * (4 * Hh) + (nb0 + 1) * 16, 4 * Hh, lane, acc1);
        }
      }
      gbar(p.bar);

      // ---- Stage 2: LSTM cell pointwise
      for (int i = gid; i < Bsz * Hh; i += nthr) {
        int b = i / Hh, j = i - b * Hh;
        const float* g = p.gates + (size_t)b * (4 * Hh);
        float gi = g[j], gf = g[Hh + j], gg = g[2 * Hh + j], go = g[3 * Hh + j];
        float ncv = sigf(gf) * p.c[i] + sigf(gi) * tanhf(gg);
        float nhv = sigf(go) * tanhf(ncv);
        p.nc[i] = ncv;
        p.nh[i] = nhv;
        p.nhb[i] = (__bf16)nhv;
      }
      gbar(p.bar);

      // ---- Stage 3: pre = tanh(enc_proj + nh @ Wdec + b_joint)  ([64,640]x[640,640])
      {
        const int NT = Jj / 16;  // 40
        for (int pair = wave; pair < 4 * (NT / 2); pair += NWAVE) {
          int rb = pair & 3, nb0 = (pair >> 2) * 2;
          const __bf16* arow = p.nhb + (size_t)(rb * 16 + m) * Hh;
          v8f acc0 = {}, acc1 = {};
          for (int kt = 0; kt < Hh / 32; ++kt) {
            v16bf a  = load_A_row(arow, kt * 32, lane);
            v16bf b0 = load_B_swz(p.wdec, kt * NT + nb0, lane);
            v16bf b1 = load_B_swz(p.wdec, kt * NT + nb0 + 1, lane);
            acc0 = WMMA_BF16(acc0, a, b0);
            acc1 = WMMA_BF16(acc1, a, b1);
          }
          int mo = (lane & 16) ? 8 : 0;
#pragma unroll
          for (int q = 0; q < 2; ++q) {
            const v8f& acc = q ? acc1 : acc0;
            int col = (nb0 + q) * 16 + m;
            float bj = p.b_joint[col];
#pragma unroll
            for (int r = 0; r < 8; ++r) {
              int row = rb * 16 + mo + r;
              float v = tanhf(acc[r] + p.encp[(size_t)row * Jj + col] + bj);
              p.preb[(size_t)row * Jj + col] = (__bf16)v;
            }
          }
        }
      }
      if (gid < Bsz) p.amax[gid] = 0ull;   // clear argmax slots for stage 4
      gbar(p.bar);

      // ---- Stage 4: logits = pre @ W_out + b_out; fused row argmax
      //      (log_softmax is monotonic -> argmax(logits) suffices)
      {
        const int NT = Vv / 16;  // 256 -> 512 pairs = exactly one pass of 512 waves
        for (int pair = wave; pair < 4 * (NT / 2); pair += NWAVE) {
          int rb = pair & 3, nb0 = (pair >> 2) * 2;
          const __bf16* arow = p.preb + (size_t)(rb * 16 + m) * Jj;
          v8f acc0 = {}, acc1 = {};
          for (int kt = 0; kt < Jj / 32; ++kt) {
            v16bf a  = load_A_row(arow, kt * 32, lane);
            v16bf b0 = load_B_swz(p.wout, kt * NT + nb0, lane);
            v16bf b1 = load_B_swz(p.wout, kt * NT + nb0 + 1, lane);
            acc0 = WMMA_BF16(acc0, a, b0);
            acc1 = WMMA_BF16(acc1, a, b1);
          }
          argmax_epilogue(acc0, rb, nb0 * 16 + m, lane, p.b_out, p.amax);
          argmax_epilogue(acc1, rb, (nb0 + 1) * 16 + m, lane, p.b_out, p.amax);
        }
      }
      gbar(p.bar);

      // ---- Stage 5: decode argmax, emit hyp, masked state update
      for (int i = gid; i < Bsz * Hh; i += nthr) {
        int b = i / Hh, j = i - b * Hh;
        unsigned long long key = p.amax[b];
        int nt = (int)(0xFFFFFFFFu - (unsigned)(key & 0xFFFFFFFFull));
        bool eq = (nt == 0);  // BLANK
        if (j == 0) {
          if (!eq) p.cur[b] = nt;
          int ef = p.endf[b] | (eq ? 1 : 0);
          p.endf[b] = ef;
          int lenf = (p.enc_len[b] <= t) || ef;
          p.out[(size_t)b * OUTW + 1 + t * MAXSYM + s] = lenf ? 0 : nt;
        }
        if (!eq) {
          p.h[i]  = p.nh[i];
          p.c[i]  = p.nc[i];
          p.hb[i] = p.nhb[i];
        }
      }
      gbar(p.bar);
    }
  }
}

// ---------------- setup kernels ---------------------------------------------

// Plain f32 -> bf16 (embedding table, used as gathered A rows)
__global__ void cvt_bf16_kernel(const float* __restrict__ src, __bf16* __restrict__ dst, int n) {
  for (int i = blockIdx.x * blockDim.x + threadIdx.x; i < n; i += gridDim.x * blockDim.x)
    dst[i] = (__bf16)src[i];
}

// f32 [K,N] row-major -> bf16 B-fragment-swizzled:
// dst[((kt*(N/16)+nt)*32 + lane)*16 + i] = src[(kt*32 + ((lane&16)?8:0) + i + (i&8))*N + nt*16 + (lane&15)]
__global__ void swz_B_kernel(const float* __restrict__ src, __bf16* __restrict__ dst,
                             int K, int N) {
  int ntiles = N >> 4;
  int total  = (K >> 5) * ntiles * 32;  // lane-units
  for (int u = blockIdx.x * blockDim.x + threadIdx.x; u < total; u += gridDim.x * blockDim.x) {
    int lane = u & 31;
    int tile = u >> 5;
    int kt = tile / ntiles, nt = tile - kt * ntiles;
    int n  = nt * 16 + (lane & 15);
    int kb = kt * 32 + ((lane & 16) ? 8 : 0);
    __bf16* d = dst + (size_t)u * 16;
    const float* sp = src + (size_t)kb * N + n;
#pragma unroll
    for (int i = 0; i < 16; ++i) d[i] = (__bf16)sp[(size_t)(i + (i & 8)) * N];
  }
}

__global__ void init_state_kernel(float* h, float* c, __bf16* hb, int* cur, int* endf,
                                  unsigned long long* amax, int* bar, int* out) {
  int i = blockIdx.x * blockDim.x + threadIdx.x;
  int n = gridDim.x * blockDim.x;
  for (int k = i; k < Bsz * Hh; k += n) { h[k] = 0.f; c[k] = 0.f; hb[k] = (__bf16)0.f; }
  for (int k = i; k < Bsz; k += n) {
    cur[k] = 0; endf[k] = 0; amax[k] = 0ull;
    out[(size_t)k * OUTW] = 0;  // reference prepends a zero column
  }
  if (i < 2) bar[i] = 0;
}

// ---------------- launch ----------------------------------------------------

extern "C" void kernel_launch(void* const* d_in, const int* in_sizes, int n_in,
                              void* d_out, int out_size, void* d_ws, size_t ws_size,
                              hipStream_t stream) {
  (void)in_sizes; (void)n_in; (void)out_size; (void)ws_size;

  const float* enc     = (const float*)d_in[0];
  const int*   enc_len = (const int*)  d_in[1];
  const float* embed_f = (const float*)d_in[2];
  const float* wx_f    = (const float*)d_in[3];
  const float* wh_f    = (const float*)d_in[4];
  const float* b_lstm  = (const float*)d_in[5];
  const float* wenc_f  = (const float*)d_in[6];
  const float* wdec_f  = (const float*)d_in[7];
  const float* b_joint = (const float*)d_in[8];
  const float* wout_f  = (const float*)d_in[9];
  const float* b_out   = (const float*)d_in[10];

  char* base = (char*)d_ws;
  size_t off = 0;
  auto carve = [&](size_t bytes) -> void* {
    void* p = base + off;
    off = (off + bytes + 255) & ~(size_t)255;
    return p;
  };

  const int nEmbed = Vv * Ee;
  const int nWx    = Ee * 4 * Hh;
  const int nWh    = Hh * 4 * Hh;
  const int nWenc  = DENCc * Jj;
  const int nWdec  = Hh * Jj;
  const int nWout  = Jj * Vv;
  const int nBH    = Bsz * Hh;

  __bf16* embed_b = (__bf16*)carve((size_t)nEmbed * 2);
  __bf16* wx_s    = (__bf16*)carve((size_t)nWx    * 2);
  __bf16* wh_s    = (__bf16*)carve((size_t)nWh    * 2);
  __bf16* wenc_s  = (__bf16*)carve((size_t)nWenc  * 2);
  __bf16* wdec_s  = (__bf16*)carve((size_t)nWdec  * 2);
  __bf16* wout_s  = (__bf16*)carve((size_t)nWout  * 2);
  __bf16* hb      = (__bf16*)carve((size_t)nBH * 2);
  __bf16* nhb     = (__bf16*)carve((size_t)nBH * 2);
  __bf16* preb    = (__bf16*)carve((size_t)nBH * 2);
  float* h     = (float*)carve((size_t)nBH * 4);
  float* c     = (float*)carve((size_t)nBH * 4);
  float* nh    = (float*)carve((size_t)nBH * 4);
  float* nc    = (float*)carve((size_t)nBH * 4);
  float* encp  = (float*)carve((size_t)nBH * 4);
  float* gates = (float*)carve((size_t)Bsz * 4 * Hh * 4);
  int* cur  = (int*)carve(Bsz * 4);
  int* endf = (int*)carve(Bsz * 4);
  unsigned long long* amax = (unsigned long long*)carve(Bsz * 8);
  int* bar  = (int*)carve(2 * 4);

  // One-time (per call) weight preparation: bf16 convert + B-fragment swizzle
  cvt_bf16_kernel<<<512, 256, 0, stream>>>(embed_f, embed_b, nEmbed);
  swz_B_kernel<<<512, 256, 0, stream>>>(wx_f,   wx_s,   Ee,    4 * Hh);
  swz_B_kernel<<<512, 256, 0, stream>>>(wh_f,   wh_s,   Hh,    4 * Hh);
  swz_B_kernel<<<512, 256, 0, stream>>>(wenc_f, wenc_s, DENCc, Jj);
  swz_B_kernel<<<512, 256, 0, stream>>>(wdec_f, wdec_s, Hh,    Jj);
  swz_B_kernel<<<512, 256, 0, stream>>>(wout_f, wout_s, Jj,    Vv);
  init_state_kernel<<<64, 256, 0, stream>>>(h, c, hb, cur, endf, amax, bar, (int*)d_out);

  Params p;
  p.enc = enc; p.enc_len = enc_len;
  p.b_lstm = b_lstm; p.b_joint = b_joint; p.b_out = b_out;
  p.embed = embed_b; p.wx = wx_s; p.wh = wh_s;
  p.wenc = wenc_s; p.wdec = wdec_s; p.wout = wout_s;
  p.h = h; p.c = c; p.nh = nh; p.nc = nc; p.encp = encp; p.gates = gates;
  p.hb = hb; p.nhb = nhb; p.preb = preb;
  p.cur = cur; p.endf = endf; p.amax = amax; p.bar = bar;
  p.out = (int*)d_out;

  rnnt_greedy_persistent<<<NBLK, NTHR, 0, stream>>>(p);
}
